// RNNStack_37632503447945
// MI455X (gfx1250) — compile-verified, matrix operand order checked by reading, not run on hardware
//
#include <hip/hip_runtime.h>
#include <hip/hip_bf16.h>

#define BB   8
#define TT   128
#define VV   32000
#define EE   128
#define HH   256
#define NSTK 20
#define KCOMB 384          // combined K: 128 (x-normed) + 256 (h)
#define WP   392           // LDS row pitch (halfs) for A / Wcomb (pad vs bank conflicts)
#define MROWS 1024         // T*B rows for the big GEMM

typedef __attribute__((ext_vector_type(16))) _Float16 v16h;
typedef __attribute__((ext_vector_type(8)))  _Float16 v8h;
typedef __attribute__((ext_vector_type(8)))  float    v8f;

// ---- WMMA helpers -----------------------------------------------------------
__device__ __forceinline__ v8f wmma_f16(v16h a, v16h b, v8f c) {
  // D = A(16x32 f16) x B(32x16 f16) + C(16x16 f32)
  return __builtin_amdgcn_wmma_f32_16x16x32_f16(false, a, false, b, (short)0, c,
                                                false, false);
}

// Load a 16x32 f16 fragment (A layout; also used for B by feeding B^T rows).
// base: row-major f16 matrix, `pitch` halfs per row, rows r0..r0+15, K chunk kc.
__device__ __forceinline__ v16h ldfrag(const _Float16* base, int pitch, int r0,
                                       int kc, int lane) {
  int r  = r0 + (lane & 15);
  int kh = kc + ((lane & 16) ? 8 : 0);
  const _Float16* p = base + (size_t)r * pitch + kh;
  v8h lo = *(const v8h*)(p);        // K = kh .. kh+7
  v8h hi = *(const v8h*)(p + 16);   // K = kh+16 .. kh+23
  v16h f;
#pragma unroll
  for (int i = 0; i < 8; ++i) { f[i] = lo[i]; f[i + 8] = hi[i]; }
  return f;
}

// ---- K0a: fold W_fc (V x 512) -> f16 (V x 256) ------------------------------
__global__ void k_fold_wfc(const float* __restrict__ wfc, _Float16* __restrict__ wfcf) {
  int idx = blockIdx.x * 256 + threadIdx.x;        // VV*HH threads
  int v = idx >> 8, k = idx & 255;
  wfcf[idx] = (_Float16)(wfc[(size_t)v * 512 + k] + wfc[(size_t)v * 512 + 256 + k]);
}

// ---- K0b: emb_cn = normalize(emb rows) -> f16 (V x 128) ---------------------
__global__ void k_embcn(const float* __restrict__ emb, _Float16* __restrict__ ecn) {
  __shared__ float red[128];
  int v = blockIdx.x, k = threadIdx.x;
  float x = emb[(size_t)v * EE + k];
  red[k] = x * x;
  __syncthreads();
  for (int s = 64; s > 0; s >>= 1) {
    if (k < s) red[k] += red[k + s];
    __syncthreads();
  }
  float rinv = 1.0f / fmaxf(sqrtf(red[0]), 1e-8f);
  ecn[(size_t)v * EE + k] = (_Float16)(x * rinv);
}

// ---- K1: sequential stack-RNN (single workgroup, WMMA recurrence) -----------
// Dynamic LDS layout (bytes):
#define A_OFF     0                       // 16 x WP f16           (12544)
#define WC_OFF    12544                   // 256 x WP f16          (200704)
#define STK_OFF   (WC_OFF + 200704)       // 8*20*128 f16          (40960)
#define HM_OFF    (STK_OFF + 40960)       // 8*256 f32             (8192)
#define XR_OFF    (HM_OFF + 8192)         // 8*128 f32             (4096)
#define POP_OFF   (XR_OFF + 4096)         // 8*128 f32             (4096)
#define OPSW_OFF  (POP_OFF + 4096)        // 5*256 f32             (5120)
#define PTR_OFF   (OPSW_OFF + 5120)       // 8*20 f32              (640)
#define NPTR_OFF  (PTR_OFF + 640)         // 8*20 f32              (640)
#define WBUF_OFF  (NPTR_OFF + 640)        // 8*20 f32              (640)
#define LGT_OFF   (WBUF_OFF + 640)        // 8*5 f32               (160)
#define P3_OFF    (LGT_OFF + 160)         // 8*3 f32               (96)
#define RINV_OFF  (P3_OFF + 96)           // 8 f32
#define PVI_OFF   (RINV_OFF + 32)         // 8 f32
#define SSI_OFF   (PVI_OFF + 32)          // 8 f32
#define K1_LDS    (SSI_OFF + 64)          // ~278 KB  (< 320 KB/WGP)

__global__ void __launch_bounds__(512) k_rnn(
    const int* __restrict__ xids, const float* __restrict__ emb,
    const float* __restrict__ Wi, const float* __restrict__ Wh,
    const float* __restrict__ Wops, const float* __restrict__ Wch,
    const float* __restrict__ sharp,
    _Float16* __restrict__ hmG, _Float16* __restrict__ pvnG,
    float* __restrict__ choiceG) {
  extern __shared__ char lds[];
  _Float16* A    = (_Float16*)(lds + A_OFF);
  _Float16* Wc   = (_Float16*)(lds + WC_OFF);
  _Float16* stk  = (_Float16*)(lds + STK_OFF);
  float* hm    = (float*)(lds + HM_OFF);
  float* xraw  = (float*)(lds + XR_OFF);
  float* pop   = (float*)(lds + POP_OFF);
  float* opsW  = (float*)(lds + OPSW_OFF);
  float* ptrL  = (float*)(lds + PTR_OFF);
  float* nptr  = (float*)(lds + NPTR_OFF);
  float* wbuf  = (float*)(lds + WBUF_OFF);
  float* lgt   = (float*)(lds + LGT_OFF);
  float* p3    = (float*)(lds + P3_OFF);
  float* rinv  = (float*)(lds + RINV_OFF);
  float* pvinv = (float*)(lds + PVI_OFF);
  float* ssinv = (float*)(lds + SSI_OFF);

  const int tid  = threadIdx.x;
  const int lane = tid & 31;
  const int wave = tid >> 5;          // 16 waves, wave w owns h columns [16w,16w+16)
  const float sh = sharp[0];

  // ---- preamble: zero A (rows 8..15 stay zero forever; h part = h0 = 0) ----
  for (int i = tid; i < 16 * WP; i += 512) A[i] = (_Float16)0.f;
  // combined weights: Wc[j][0..127]=W_i[j], Wc[j][128..383]=W_h[j]   (f16)
  for (int i = tid; i < HH * KCOMB; i += 512) {
    int j = i / KCOMB, k = i % KCOMB;
    float w = (k < EE) ? Wi[j * EE + k] : Wh[j * HH + (k - EE)];
    Wc[j * WP + k] = (_Float16)w;
  }
  // folded small heads: rows 0..2 = W_ops, 3..4 = W_choose
  for (int i = tid; i < 5 * HH; i += 512) {
    int q = i / HH, k = i % HH;
    float w = (q < 3) ? (Wops[q * 512 + k] + Wops[q * 512 + 256 + k])
                      : (Wch[(q - 3) * 512 + k] + Wch[(q - 3) * 512 + 256 + k]);
    opsW[i] = w;
  }
  for (int i = tid; i < BB * NSTK * EE; i += 512) stk[i] = (_Float16)0.001f;
  for (int i = tid; i < BB * NSTK; i += 512) ptrL[i] = ((i % NSTK) == 0) ? 1.f : 0.f;
  __syncthreads();

  for (int t = 0; t < TT; ++t) {
    // -- gather x_t --
    for (int i = tid; i < BB * EE; i += 512) {
      int b = i >> 7, k = i & 127;
      xraw[i] = emb[(size_t)xids[b * TT + t] * EE + k];
    }
    __syncthreads();
    if (tid < BB) {
      float ss = 0.f;
      for (int k = 0; k < EE; ++k) { float v = xraw[tid * EE + k]; ss += v * v; }
      rinv[tid] = 1.f / fmaxf(sqrtf(ss), 1e-12f);
    }
    __syncthreads();
    for (int i = tid; i < BB * EE; i += 512) {
      int b = i >> 7, k = i & 127;
      A[b * WP + k] = (_Float16)(xraw[i] * rinv[b]);   // normalized x into A[:,0:128]
    }
    __syncthreads();

    // -- hnew = tanh([xn h] @ Wc^T) via WMMA (16 waves x one 16-col tile) --
    v8f acc = {0.f, 0.f, 0.f, 0.f, 0.f, 0.f, 0.f, 0.f};
    int n0 = wave * 16;
#pragma unroll
    for (int kc = 0; kc < KCOMB; kc += 32) {
      v16h af = ldfrag(A, WP, 0, kc, lane);
      v16h bf = ldfrag(Wc, WP, n0, kc, lane);
      acc = wmma_f16(af, bf, acc);
    }
    __syncthreads();                          // everyone done READING A's h-part
    if (lane < 16) {
#pragma unroll
      for (int r = 0; r < 8; ++r) {           // rows 0..7 == batch b (rows 8..15 were 0)
        float hv  = tanhf(acc[r]);
        float hmv = 1.f - fmaxf(hv, 0.f);     // hm = 1 - relu(h)
        int b = r, j = n0 + lane;
        A[b * WP + EE + j] = (_Float16)hv;    // recurrent state for next step
        hm[b * HH + j] = hmv;
        hmG[(size_t)(t * BB + b) * HH + j] = (_Float16)hmv;
      }
    }
    __syncthreads();

    // -- pop_val (old ptr, old stack) + small head logits --
    for (int i = tid; i < BB * EE; i += 512) {
      int b = i >> 7, e = i & 127;
      float a = 0.f;
      for (int n = 0; n < NSTK; ++n)
        a += ptrL[b * NSTK + n] * (float)stk[(b * NSTK + n) * EE + e];
      pop[i] = a;
    }
    if (tid < 40) {
      int b = tid & 7, q = tid >> 3;
      float a = 0.f;
      for (int k = 0; k < HH; ++k) a += hm[b * HH + k] * opsW[q * HH + k];
      lgt[b * 5 + q] = a;
    }
    __syncthreads();

    // -- softmaxes + pop_val norm --
    if (tid < BB) {
      int b = tid;
      float l0 = lgt[b*5+0], l1 = lgt[b*5+1], l2 = lgt[b*5+2];
      float mx = fmaxf(l0, fmaxf(l1, l2));
      float e0 = __expf(l0 - mx), e1 = __expf(l1 - mx), e2 = __expf(l2 - mx);
      float si = 1.f / (e0 + e1 + e2);
      p3[b*3+0] = e0 * si; p3[b*3+1] = e1 * si; p3[b*3+2] = e2 * si;
      float c0l = lgt[b*5+3], c1l = lgt[b*5+4];
      float cm = fmaxf(c0l, c1l);
      float ce0 = __expf(c0l - cm), ce1 = __expf(c1l - cm);
      float ci = 1.f / (ce0 + ce1);
      choiceG[(t * BB + b) * 2 + 0] = ce0 * ci;
      choiceG[(t * BB + b) * 2 + 1] = ce1 * ci;
      float ss = 0.f;
      for (int e = 0; e < EE; ++e) { float v = pop[b * EE + e]; ss += v * v; }
      pvinv[b] = 1.f / fmaxf(sqrtf(ss), 1e-8f);
    }
    __syncthreads();

    // -- new ptr candidates, push weights; emit normalized pop_val --
    if (tid < BB * NSTK) {
      int b = tid / NSTK, n = tid % NSTK;
      float pps = ptrL[b * NSTK + (n + NSTK - 1) % NSTK];  // roll(+1)
      float ppo = ptrL[b * NSTK + (n + 1) % NSTK];         // roll(-1)
      float pu = p3[b*3+0], po = p3[b*3+1], nn = p3[b*3+2];
      float np = pu * pps + po * ppo + nn * ptrL[b * NSTK + n];
      nptr[tid] = powf(fmaxf(np, 1e-8f), sh);
      wbuf[tid] = pu * pps;                                // w = push * p_push
    }
    for (int i = tid; i < BB * EE; i += 512) {
      int b = i >> 7, e = i & 127;
      pvnG[(size_t)(t * BB + b) * EE + e] = (_Float16)(pop[i] * pvinv[b]);
    }
    __syncthreads();

    // -- stack blend (raw x!) + ptr sum --
    for (int i = tid; i < BB * NSTK * EE; i += 512) {
      int e = i & 127, bn = i >> 7, b = bn / NSTK;
      float w = wbuf[bn];
      stk[i] = (_Float16)((float)stk[i] * (1.f - w) + w * xraw[b * EE + e]);
    }
    if (tid < BB) {
      float s = 0.f;
      for (int n = 0; n < NSTK; ++n) s += nptr[tid * NSTK + n];
      ssinv[tid] = 1.f / s;
    }
    __syncthreads();
    if (tid < BB * NSTK) ptrL[tid] = nptr[tid] * ssinv[tid / NSTK];
    __syncthreads();
  }
}

// ---- K2: big WMMA GEMM: logits L -> ws, out = c1 * syntactic ----------------
__global__ void __launch_bounds__(256) k_gemm(
    const _Float16* __restrict__ hmG, const _Float16* __restrict__ wfcf,
    const _Float16* __restrict__ pvnG, const _Float16* __restrict__ ecn,
    const float* __restrict__ choiceG, float* __restrict__ Lws,
    float* __restrict__ out) {
  int lane = threadIdx.x & 31;
  int wave = threadIdx.x >> 5;
  int m0 = blockIdx.y * 16;                    // 64 m-tiles over 1024 rows
  int n0 = (blockIdx.x * 8 + wave) * 16;       // 2000 n-tiles over 32000 cols
  v8f accL = {0.f, 0.f, 0.f, 0.f, 0.f, 0.f, 0.f, 0.f};
  v8f accS = {0.f, 0.f, 0.f, 0.f, 0.f, 0.f, 0.f, 0.f};
#pragma unroll
  for (int kc = 0; kc < HH; kc += 32) {        // semantic logits: K=256
    v16h a = ldfrag(hmG, HH, m0, kc, lane);
    v16h b = ldfrag(wfcf, HH, n0, kc, lane);
    accL = wmma_f16(a, b, accL);
  }
#pragma unroll
  for (int kc = 0; kc < EE; kc += 32) {        // syntactic: K=128
    v16h a = ldfrag(pvnG, EE, m0, kc, lane);
    v16h b = ldfrag(ecn, EE, n0, kc, lane);
    accS = wmma_f16(a, b, accS);
  }
  int col   = n0 + (lane & 15);
  int rbase = m0 + ((lane & 16) ? 8 : 0);
#pragma unroll
  for (int r = 0; r < 8; ++r) {
    int m = rbase + r;                         // m = t*8 + b
    Lws[(size_t)m * VV + col] = accL[r];
    int t = m >> 3, b = m & 7;
    float c1 = choiceG[m * 2 + 1];
    out[((size_t)(b * TT + t)) * VV + col] = c1 * accS[r];   // out is (B,T,V)
  }
}

// ---- K3: per-row online softmax stats over V ----------------------------
__global__ void k_stats(const float* __restrict__ Lws, float* __restrict__ rmax,
                        float* __restrict__ rsum) {
  __shared__ float sm[256], ss[256];
  int m = blockIdx.x, tid = threadIdx.x;
  const float* row = Lws + (size_t)m * VV;
  float mx = -3.4e38f, s = 0.f;
  for (int v = tid; v < VV; v += 256) {
    float x = row[v];
    if (x > mx) { s *= __expf(mx - x); mx = x; }
    s += __expf(x - mx);
  }
  sm[tid] = mx; ss[tid] = s;
  __syncthreads();
  for (int st = 128; st > 0; st >>= 1) {
    if (tid < st) {
      float m1 = sm[tid], m2 = sm[tid + st];
      float M = fmaxf(m1, m2);
      ss[tid] = ss[tid] * __expf(m1 - M) + ss[tid + st] * __expf(m2 - M);
      sm[tid] = M;
    }
    __syncthreads();
  }
  if (tid == 0) { rmax[m] = sm[0]; rsum[m] = ss[0]; }
}

// ---- K4: out += c0 * softmax(L) ------------------------------------------
__global__ void k_combine(const float* __restrict__ Lws,
                          const float* __restrict__ choiceG,
                          const float* __restrict__ rmax,
                          const float* __restrict__ rsum,
                          float* __restrict__ out) {
  int m = blockIdx.x, tid = threadIdx.x;
  float c0 = choiceG[m * 2 + 0];
  float M = rmax[m], Si = 1.f / rsum[m];
  int t = m >> 3, b = m & 7;
  const float* row = Lws + (size_t)m * VV;
  float* orow = out + ((size_t)(b * TT + t)) * VV;
  for (int v = tid; v < VV; v += 256)
    orow[v] += c0 * __expf(row[v] - M) * Si;
}

// ---- host ------------------------------------------------------------------
extern "C" void kernel_launch(void* const* d_in, const int* in_sizes, int n_in,
                              void* d_out, int out_size, void* d_ws, size_t ws_size,
                              hipStream_t stream) {
  (void)in_sizes; (void)n_in; (void)out_size; (void)ws_size;
  const int*   xids  = (const int*)d_in[0];
  const float* emb   = (const float*)d_in[1];
  const float* Wi    = (const float*)d_in[2];
  const float* Wh    = (const float*)d_in[3];
  const float* Wfc   = (const float*)d_in[4];
  const float* Wops  = (const float*)d_in[5];
  const float* Wch   = (const float*)d_in[6];
  const float* sharp = (const float*)d_in[7];
  float* out = (float*)d_out;

  char* ws = (char*)d_ws;
  size_t off = 0;
  float*    Lws  = (float*)(ws + off);    off += (size_t)MROWS * VV * 4;  // 131.07 MB
  _Float16* wfcf = (_Float16*)(ws + off); off += (size_t)VV * HH * 2;     // 16.38 MB
  _Float16* ecn  = (_Float16*)(ws + off); off += (size_t)VV * EE * 2;     // 8.19 MB
  _Float16* hmG  = (_Float16*)(ws + off); off += (size_t)MROWS * HH * 2;  // 512 KB
  _Float16* pvnG = (_Float16*)(ws + off); off += (size_t)MROWS * EE * 2;  // 256 KB
  float* choiceG = (float*)(ws + off);    off += (size_t)MROWS * 2 * 4;
  float* rmax    = (float*)(ws + off);    off += (size_t)MROWS * 4;
  float* rsum    = (float*)(ws + off);    off += (size_t)MROWS * 4;

  k_fold_wfc<<<dim3(VV), dim3(256), 0, stream>>>(Wfc, wfcf);
  k_embcn<<<dim3(VV), dim3(128), 0, stream>>>(emb, ecn);
  k_rnn<<<dim3(1), dim3(512), K1_LDS, stream>>>(xids, emb, Wi, Wh, Wops, Wch,
                                                sharp, hmG, pvnG, choiceG);
  k_gemm<<<dim3(250, 64), dim3(256), 0, stream>>>(hmG, wfcf, pvnG, ecn, choiceG,
                                                  Lws, out);
  k_stats<<<dim3(MROWS), dim3(256), 0, stream>>>(Lws, rmax, rsum);
  k_combine<<<dim3(MROWS), dim3(256), 0, stream>>>(Lws, choiceG, rmax, rsum, out);
}
